// WindowAttention_35003983462720
// MI455X (gfx1250) — matrix operand order repaired and checked by reading, not compiled, not run on hardware
//
#include <hip/hip_runtime.h>
#include <hip/hip_bf16.h>
#include <cstdint>
#include <cstddef>

// ---------------- CDNA5 WMMA types ----------------
typedef __attribute__((ext_vector_type(16))) _Float16 v16h;
typedef __attribute__((ext_vector_type(8)))  float    v8f;

#define NTOK 262144   // B*H*W == Nwin*64  (16*128*128)
#define CDIM 256
#define FFD  512
#define QKVD 768

union Frag16 { v16h v; uint4 u[2]; };

// A-matrix 16x32 f16 fragment (ISA 7.12.2): lane half=lane>>4, row=lane&15.
// elements 0..15 -> K = k0 + 16*half + (0..15)  (contiguous, matches B)
__device__ __forceinline__ v16h load_a_frag(const _Float16* rowptr, int k0, int hf) {
  Frag16 f;
  f.u[0] = *reinterpret_cast<const uint4*>(rowptr + k0 + 16 * hf);
  f.u[1] = *reinterpret_cast<const uint4*>(rowptr + k0 + 16 * hf + 8);
  return f.v;
}
// B-matrix 32x16 f16 fragment: lane holds column n=lane&15,
// elements 0..15 -> K = k0 + 16*half + (0..15)  (contiguous 32B)
__device__ __forceinline__ v16h load_b_frag(const _Float16* rowptr, int k0, int hf) {
  return *reinterpret_cast<const v16h*>(rowptr + k0 + 16 * hf);
}

// ------------- WMMA GEMM v2: Y = act(X @ W^T + b) (+resid) (+fused row-LN) -------------
// X: [M,K] f16 row-major; W: [N,K] f16 row-major.
// Block = 256 threads (8 wave32), tile BM=64 x BN=256.
// Wave w: row strip (w>>1)*16, column half (w&1)*128 -> 8 accumulators.
// Weight k-slab (256 rows x 32) staged in LDS per k-step, shared by all 8 waves.
// DOLN: epilogue computes LN over the full 256-wide row (requires N==256, gridDim.y==1):
//       out = LN(resid + gemm + bias) * gamma + beta   (post-norm residual, fused)
template <int ACT, bool DOLN>
__global__ __launch_bounds__(256)
void k_gemm2(const _Float16* __restrict__ A, const _Float16* __restrict__ W,
             const float* __restrict__ bias, const float* __restrict__ resid,
             const float* __restrict__ gamma, const float* __restrict__ beta,
             float* __restrict__ out32, _Float16* __restrict__ out16,
             int M, int N, int K) {
  __shared__ union { _Float16 b[256][32]; float f[64][256]; } sm;  // 64KB, reused
  __shared__ float smu[64], srs[64];
  __shared__ uint4 s_abuf[8];   // landing pad for the async warm-up load

  const int tid   = threadIdx.x;
  const int wv    = tid >> 5;
  const int lane  = tid & 31;
  const int hf    = lane >> 4;
  const int lr    = lane & 15;
  const int strip = wv >> 1;      // 0..3 (16-row strips)
  const int ch    = wv & 1;       // column half (128 cols)
  const int m0    = blockIdx.x * 64;
  const int n0    = blockIdx.y * 256;
  const _Float16* arow = A + (size_t)(m0 + strip * 16 + lr) * K;

  // gfx1250 async warm-up: pull the first weight cacheline toward LDS/L2.
  // Correctness-neutral: lands only in s_abuf (or is dropped if out of range).
  {
    unsigned dsa  = (unsigned)(size_t)(&s_abuf[0]);
    unsigned voff = 0u;
    const _Float16* wbase = W + (size_t)n0 * K;
    asm volatile("global_load_async_to_lds_b128 %0, %1, %2 offset:0"
                 :: "v"(dsa), "v"(voff), "s"(wbase) : "memory");
  }

  v8f acc[8] = {};
  for (int k0 = 0; k0 < K; k0 += 32) {
    // cooperatively stage the 256x32 weight slab (thread t owns W row n0+t)
    const _Float16* wr = W + (size_t)(n0 + tid) * K + k0;
    uint4 w0 = *reinterpret_cast<const uint4*>(wr);
    uint4 w1 = *reinterpret_cast<const uint4*>(wr + 8);
    uint4 w2 = *reinterpret_cast<const uint4*>(wr + 16);
    uint4 w3 = *reinterpret_cast<const uint4*>(wr + 24);
    __syncthreads();                              // prior slab fully consumed
    uint4* dst = reinterpret_cast<uint4*>(&sm.b[tid][0]);
    dst[0] = w0; dst[1] = w1; dst[2] = w2; dst[3] = w3;
    __syncthreads();

    if (k0 + 32 < K) __builtin_prefetch(arow + k0 + 32, 0, 0);
    v16h a = load_a_frag(arow, k0, hf);
#pragma unroll
    for (int nt = 0; nt < 8; ++nt) {
      v16h b = *reinterpret_cast<const v16h*>(&sm.b[ch * 128 + nt * 16 + lr][16 * hf]);
      acc[nt] = __builtin_amdgcn_wmma_f32_16x16x32_f16(
          false, a, false, b, (short)0, acc[nt], false, false);
    }
  }
  asm volatile("s_wait_asynccnt 0x0" ::: "memory");

  if (!DOLN) {
#pragma unroll
    for (int nt = 0; nt < 8; ++nt) {
      const int col = n0 + ch * 128 + nt * 16 + lr;
      const float bb = bias ? bias[col] : 0.f;
#pragma unroll
      for (int r = 0; r < 8; ++r) {
        const int row = m0 + strip * 16 + hf * 8 + r;
        float v = acc[nt][r] + bb;
        if (ACT == 1) v = fmaxf(v, 0.f);
        if (ACT == 2) v = 0.5f * v * (1.f + erff(v * 0.70710678118654752f));
        const size_t o = (size_t)row * N + col;
        if (resid) v += resid[o];
        if (out32) out32[o] = v;
        if (out16) out16[o] = (_Float16)v;
      }
    }
  } else {
    // fused post-norm residual LN over the full 256-wide row
    __syncthreads();   // done with sm.b
#pragma unroll
    for (int nt = 0; nt < 8; ++nt) {
      const int colL = ch * 128 + nt * 16 + lr;
      const float bb = bias[n0 + colL];
#pragma unroll
      for (int r = 0; r < 8; ++r) {
        const int rowL = strip * 16 + hf * 8 + r;
        sm.f[rowL][colL] = acc[nt][r] + bb + resid[(size_t)(m0 + rowL) * N + n0 + colL];
      }
    }
    __syncthreads();
    if (tid < 64) {
      float s = 0.f;
      for (int j = 0; j < 256; ++j) s += sm.f[tid][j];
      const float mu = s * (1.f / 256.f);
      float vs = 0.f;
      for (int j = 0; j < 256; ++j) { const float d = sm.f[tid][j] - mu; vs += d * d; }
      smu[tid] = mu;
      srs[tid] = rsqrtf(vs * (1.f / 256.f) + 1e-5f);
    }
    __syncthreads();
    for (int i = tid; i < 64 * 256; i += 256) {
      const int rowL = i >> 8, col = i & 255;
      const float xn = (sm.f[rowL][col] - smu[rowL]) * srs[rowL] * gamma[col] + beta[col];
      const size_t o = (size_t)(m0 + rowL) * N + col;
      out32[o] = xn;
      out16[o] = (_Float16)xn;
    }
  }
}

// ---------------- window attention: one block per (window, head) ----------------
__global__ __launch_bounds__(128)
void k_attn(const _Float16* __restrict__ qkv, _Float16* __restrict__ aout, float scale) {
  __shared__ _Float16 sq[64][32];
  __shared__ _Float16 sk[64][32];
  __shared__ _Float16 svT[32][64];
  __shared__ float    ssc[64][64];
  __shared__ _Float16 satt[64][64];

  const int win = blockIdx.x, head = blockIdx.y;
  const int tid = threadIdx.x;
  const _Float16* base = qkv + (size_t)win * 64 * QKVD + head * 32;

  for (int i = tid; i < 64 * 32; i += 128) {
    const int t = i >> 5, d = i & 31;
    const _Float16* rp = base + (size_t)t * QKVD;
    sq[t][d]  = rp[d];
    sk[t][d]  = rp[CDIM + d];
    svT[d][t] = rp[2 * CDIM + d];   // stage V transposed for contiguous B-frags
  }
  __syncthreads();

  const int wv = tid >> 5, lane = tid & 31, hf = lane >> 4, lr = lane & 15;

  // scores = q @ k^T * scale   (K = hd = 32 -> exactly one WMMA per tile)
  {
    v16h a = load_a_frag(&sq[wv * 16 + lr][0], 0, hf);
#pragma unroll
    for (int nt = 0; nt < 4; ++nt) {
      v16h b = load_b_frag(&sk[nt * 16 + lr][0], 0, hf);
      v8f c = {};
      c = __builtin_amdgcn_wmma_f32_16x16x32_f16(false, a, false, b, (short)0, c, false, false);
#pragma unroll
      for (int r = 0; r < 8; ++r)
        ssc[wv * 16 + hf * 8 + r][nt * 16 + lr] = c[r] * scale;
    }
  }
  __syncthreads();

  // softmax over rows (64 rows, one thread each)
  if (tid < 64) {
    float mx = -3.4e38f;
    for (int j = 0; j < 64; ++j) mx = fmaxf(mx, ssc[tid][j]);
    float s = 0.f;
    for (int j = 0; j < 64; ++j) { float e = __expf(ssc[tid][j] - mx); ssc[tid][j] = e; s += e; }
    const float inv = 1.f / s;
    for (int j = 0; j < 64; ++j) satt[tid][j] = (_Float16)(ssc[tid][j] * inv);
  }
  __syncthreads();

  // o = att @ v  (K = 64 -> two WMMA k-steps), heads concatenated in aout
#pragma unroll
  for (int nt = 0; nt < 2; ++nt) {
    v8f c = {};
#pragma unroll
    for (int ks = 0; ks < 2; ++ks) {
      v16h a = load_a_frag(&satt[wv * 16 + lr][0], ks * 32, hf);
      v16h b = load_b_frag(&svT[nt * 16 + lr][0], ks * 32, hf);
      c = __builtin_amdgcn_wmma_f32_16x16x32_f16(false, a, false, b, (short)0, c, false, false);
    }
#pragma unroll
    for (int r = 0; r < 8; ++r) {
      const size_t row = (size_t)win * 64 + wv * 16 + hf * 8 + r;
      aout[row * CDIM + head * 32 + nt * 16 + lr] = (_Float16)c[r];
    }
  }
}

// ---------------- elementwise / LN / layout kernels ----------------
__device__ __forceinline__ float blk_sum256(float v, float* sb) {
  const int t = threadIdx.x;
  sb[t] = v; __syncthreads();
  for (int s = 128; s > 0; s >>= 1) { if (t < s) sb[t] += sb[t + s]; __syncthreads(); }
  const float r = sb[0]; __syncthreads();
  return r;
}

__global__ void k_f32_to_f16(const float* __restrict__ s, _Float16* __restrict__ d, size_t n) {
  size_t i = (size_t)blockIdx.x * blockDim.x + threadIdx.x;
  if (i < n) d[i] = (_Float16)s[i];
}

__global__ void k_nchw2nhwc(const float* __restrict__ x, float* __restrict__ xh) {
  size_t idx = (size_t)blockIdx.x * blockDim.x + threadIdx.x;
  const size_t total = (size_t)16 * CDIM * 16384;
  if (idx >= total) return;
  const size_t b = idx / ((size_t)CDIM * 16384);
  const size_t rem = idx % ((size_t)CDIM * 16384);
  const size_t c = rem / 16384, hw = rem % 16384;
  xh[((size_t)b * 16384 + hw) * CDIM + c] = x[idx];
}

// LN1 + window partition + pos_emb  -> w (f32 + f16), row = window-major token order
__global__ __launch_bounds__(256)
void k_ln_part(const float* __restrict__ xh, const float* __restrict__ g,
               const float* __restrict__ be, const float* __restrict__ pos,
               float* __restrict__ w32, _Float16* __restrict__ w16) {
  __shared__ float sb[256];
  const int bhw = blockIdx.x, c = threadIdx.x;
  const float v = xh[(size_t)bhw * CDIM + c];
  const float mu = blk_sum256(v, sb) * (1.f / 256.f);
  const float dd = v - mu;
  const float var = blk_sum256(dd * dd, sb) * (1.f / 256.f);
  const int b = bhw >> 14, hw = bhw & 16383, h = hw >> 7, wd = hw & 127;
  const int ph = h >> 3, wi = h & 7, pw = wd >> 3, wj = wd & 7;
  const int tin = wi * 8 + wj;
  const size_t row = ((size_t)((b * 16 + ph) * 16 + pw)) * 64 + tin;
  const float xn = dd * rsqrtf(var + 1e-5f) * g[c] + be[c] + pos[(size_t)tin * CDIM + c];
  w32[row * CDIM + c] = xn;
  w16[row * CDIM + c] = (_Float16)xn;
}

// window reverse + residual: y(=xh in place) += w ; z16 = LN(y, norm2) in f16
__global__ __launch_bounds__(256)
void k_unpart_ln2(const float* __restrict__ w32, float* __restrict__ y,
                  const float* __restrict__ g, const float* __restrict__ be,
                  _Float16* __restrict__ z16) {
  __shared__ float sb[256];
  const int bhw = blockIdx.x, c = threadIdx.x;
  const int b = bhw >> 14, hw = bhw & 16383, h = hw >> 7, wd = hw & 127;
  const int ph = h >> 3, wi = h & 7, pw = wd >> 3, wj = wd & 7;
  const size_t row = ((size_t)((b * 16 + ph) * 16 + pw)) * 64 + wi * 8 + wj;
  const float v = w32[row * CDIM + c] + y[(size_t)bhw * CDIM + c];
  y[(size_t)bhw * CDIM + c] = v;
  const float mu = blk_sum256(v, sb) * (1.f / 256.f);
  const float dd = v - mu;
  const float var = blk_sum256(dd * dd, sb) * (1.f / 256.f);
  z16[(size_t)bhw * CDIM + c] = (_Float16)(dd * rsqrtf(var + 1e-5f) * g[c] + be[c]);
}

// 3x3 depthwise conv (SAME) + bias + gelu, NHWC f16
__global__ __launch_bounds__(256)
void k_dwconv(const _Float16* __restrict__ g1, const float* __restrict__ kw,
              const float* __restrict__ kb, _Float16* __restrict__ g2) {
  const int bhw = blockIdx.x;
  const int b = bhw >> 14, hw = bhw & 16383, h = hw >> 7, wd = hw & 127;
  for (int f = threadIdx.x; f < FFD; f += 256) {
    float s = kb[f];
#pragma unroll
    for (int ky = 0; ky < 3; ++ky) {
      const int hy = h + ky - 1; if (hy < 0 || hy > 127) continue;
#pragma unroll
      for (int kx = 0; kx < 3; ++kx) {
        const int wx = wd + kx - 1; if (wx < 0 || wx > 127) continue;
        const size_t src = (((size_t)b * 128 + hy) * 128 + wx) * FFD + f;
        s += (float)g1[src] * kw[(size_t)(ky * 3 + kx) * FFD + f];
      }
    }
    s = 0.5f * s * (1.f + erff(s * 0.70710678118654752f));
    g2[(size_t)bhw * FFD + f] = (_Float16)s;
  }
}

__global__ void k_out_tr(const float* __restrict__ src, float* __restrict__ dst) {
  size_t idx = (size_t)blockIdx.x * blockDim.x + threadIdx.x;
  const size_t total = (size_t)16 * CDIM * 16384;
  if (idx >= total) return;
  const size_t b = idx / ((size_t)CDIM * 16384);
  const size_t rem = idx % ((size_t)CDIM * 16384);
  const size_t c = rem / 16384, hw = rem % 16384;
  dst[idx] = src[((size_t)b * 16384 + hw) * CDIM + c];
}

// ---------------- host launcher ----------------
extern "C" void kernel_launch(void* const* d_in, const int* in_sizes, int n_in,
                              void* d_out, int out_size, void* d_ws, size_t ws_size,
                              hipStream_t stream) {
  (void)in_sizes; (void)n_in; (void)out_size; (void)ws_size;
  const float* x        = (const float*)d_in[0];
  const float* pos_emb  = (const float*)d_in[1];
  const float* norm1_w  = (const float*)d_in[2];
  const float* norm1_b  = (const float*)d_in[3];
  const float* qkv_w    = (const float*)d_in[4];
  const float* qkv_b    = (const float*)d_in[5];
  const float* outp_w   = (const float*)d_in[6];
  const float* outp_b   = (const float*)d_in[7];
  const float* ff1_w    = (const float*)d_in[8];
  const float* ff1_b    = (const float*)d_in[9];
  const float* ff2_w    = (const float*)d_in[10];
  const float* ff2_b    = (const float*)d_in[11];
  const float* ln1_w    = (const float*)d_in[12];
  const float* ln1_b    = (const float*)d_in[13];
  const float* ln2_w    = (const float*)d_in[14];
  const float* ln2_b    = (const float*)d_in[15];
  const float* norm2_w  = (const float*)d_in[16];
  const float* norm2_b  = (const float*)d_in[17];
  const float* lin1_w   = (const float*)d_in[18];
  const float* lin1_b   = (const float*)d_in[19];
  const float* dw_w     = (const float*)d_in[20];
  const float* dw_b     = (const float*)d_in[21];
  const float* lin2_w   = (const float*)d_in[22];
  const float* lin2_b   = (const float*)d_in[23];
  const float* out_w    = (const float*)d_in[24];
  const float* out_b    = (const float*)d_in[25];
  float* dout = (float*)d_out;

  // workspace bump allocator
  char* ws = (char*)d_ws; size_t off = 0;
  auto alloc = [&](size_t bytes) -> void* {
    void* p = ws + off; off += (bytes + 255) & ~(size_t)255; return p;
  };
  float*     wf32   = (float*)alloc((size_t)NTOK * CDIM * 4);
  float*     xh     = (float*)alloc((size_t)NTOK * CDIM * 4);   // becomes y in place
  _Float16*  wf16   = (_Float16*)alloc((size_t)NTOK * CDIM * 2);
  _Float16*  qkv16  = (_Float16*)alloc((size_t)NTOK * QKVD * 2); // reused as g2
  _Float16*  attn16 = (_Float16*)alloc((size_t)NTOK * CDIM * 2); // reused as yh16
  _Float16*  ff16   = (_Float16*)alloc((size_t)NTOK * FFD * 2);  // reused as g1
  _Float16*  wqkv16 = (_Float16*)alloc((size_t)2 * QKVD * CDIM * 2);
  _Float16*  wout16 = (_Float16*)alloc((size_t)2 * CDIM * CDIM * 2);
  _Float16*  wff116 = (_Float16*)alloc((size_t)2 * FFD * CDIM * 2);
  _Float16*  wff216 = (_Float16*)alloc((size_t)2 * CDIM * FFD * 2);
  _Float16*  wl116  = (_Float16*)alloc((size_t)FFD * CDIM * 2);
  _Float16*  wl216  = (_Float16*)alloc((size_t)CDIM * FFD * 2);
  _Float16*  wow16  = (_Float16*)alloc((size_t)CDIM * CDIM * 2);

  auto cvt = [&](const float* s, _Float16* d, size_t n) {
    k_f32_to_f16<<<(unsigned)((n + 255) / 256), 256, 0, stream>>>(s, d, n);
  };
  cvt(qkv_w,  wqkv16, (size_t)2 * QKVD * CDIM);
  cvt(outp_w, wout16, (size_t)2 * CDIM * CDIM);
  cvt(ff1_w,  wff116, (size_t)2 * FFD * CDIM);
  cvt(ff2_w,  wff216, (size_t)2 * CDIM * FFD);
  cvt(lin1_w, wl116,  (size_t)FFD * CDIM);
  cvt(lin2_w, wl216,  (size_t)CDIM * FFD);
  cvt(out_w,  wow16,  (size_t)CDIM * CDIM);

  const size_t total = (size_t)16 * CDIM * 16384;
  k_nchw2nhwc<<<(unsigned)((total + 255) / 256), 256, 0, stream>>>(x, xh);
  k_ln_part<<<NTOK, 256, 0, stream>>>(xh, norm1_w, norm1_b, pos_emb, wf32, wf16);

  const dim3 grid_qkv(NTOK / 64, QKVD / 256);
  const dim3 grid_c(NTOK / 64, 1);
  const dim3 grid_ff(NTOK / 64, FFD / 256);
  const float scale = 0.17677669529663687f;  // 1/sqrt(32)

  for (int l = 0; l < 2; ++l) {
    const _Float16* Wq = wqkv16 + (size_t)l * QKVD * CDIM;
    const _Float16* Wo = wout16 + (size_t)l * CDIM * CDIM;
    const _Float16* W1 = wff116 + (size_t)l * FFD * CDIM;
    const _Float16* W2 = wff216 + (size_t)l * CDIM * FFD;
    // qkv = w @ Wqkv^T + b  (f16 only)
    k_gemm2<0, false><<<grid_qkv, 256, 0, stream>>>(wf16, Wq, qkv_b + (size_t)l * QKVD,
        nullptr, nullptr, nullptr, nullptr, qkv16, NTOK, QKVD, CDIM);
    // per-(window,head) attention
    k_attn<<<dim3(NTOK / 64, 8), 128, 0, stream>>>(qkv16, attn16, scale);
    // w = LN(w + attn @ Wo^T + b)  -- out projection with FUSED residual LayerNorm
    k_gemm2<0, true><<<grid_c, 256, 0, stream>>>(attn16, Wo, outp_b + (size_t)l * CDIM,
        wf32, ln1_w + (size_t)l * CDIM, ln1_b + (size_t)l * CDIM,
        wf32, wf16, NTOK, CDIM, CDIM);
    // ff = relu(w @ W1^T + b1)
    k_gemm2<1, false><<<grid_ff, 256, 0, stream>>>(wf16, W1, ff1_b + (size_t)l * FFD,
        nullptr, nullptr, nullptr, nullptr, ff16, NTOK, FFD, CDIM);
    // w = LN(w + ff @ W2^T + b2)  -- FUSED residual LayerNorm
    k_gemm2<0, true><<<grid_c, 256, 0, stream>>>(ff16, W2, ff2_b + (size_t)l * CDIM,
        wf32, ln2_w + (size_t)l * CDIM, ln2_b + (size_t)l * CDIM,
        wf32, wf16, NTOK, CDIM, FFD);
  }

  // y = window_reverse(w) + xh (in place into xh); zh16 = LN(y, norm2)
  _Float16* zh16 = wf16;  // reuse
  k_unpart_ln2<<<NTOK, 256, 0, stream>>>(wf32, xh, norm2_w, norm2_b, zh16);

  // g1 = gelu(z @ lin1^T + b)
  _Float16* g1 = ff16;
  k_gemm2<2, false><<<grid_ff, 256, 0, stream>>>(zh16, wl116, lin1_b,
      nullptr, nullptr, nullptr, nullptr, g1, NTOK, FFD, CDIM);
  // g2 = gelu(dwconv3x3(g1) + dw_b)
  _Float16* g2 = qkv16;  // reuse (384MB >= 256MB)
  k_dwconv<<<NTOK, 256, 0, stream>>>(g1, dw_w, dw_b, g2);
  // y += g2 @ lin2^T + b ; keep f16 copy for final projection
  _Float16* yh16 = attn16;  // reuse
  k_gemm2<0, false><<<grid_c, 256, 0, stream>>>(g2, wl216, lin2_b,
      xh, nullptr, nullptr, nullptr, yh16, NTOK, CDIM, FFD);
  // out = y @ out_w^T + out_b (NHWC f32 into wf32), then transpose to NCHW
  k_gemm2<0, false><<<grid_c, 256, 0, stream>>>(yh16, wow16, out_b,
      nullptr, nullptr, nullptr, wf32, nullptr, NTOK, CDIM, CDIM);
  k_out_tr<<<(unsigned)((total + 255) / 256), 256, 0, stream>>>(wf32, dout);
}
